// flash_attn1_678604832927
// MI455X (gfx1250) — matrix operand order repaired and checked by reading, not compiled, not run on hardware
//
#include <hip/hip_runtime.h>
#include <math.h>
#include <stdint.h>

#define S_LEN    32768
#define HD       256
#define BC       64
#define ROWS_WG  128
#define NTHREADS 256
#define KS_STRIDE 264   // halfs per 256-wide row (16B multiple, bank-spread pad)
#define VT_STRIDE 72    // halfs per Vt row (64 keys + 8 pad)
#define NITER    (S_LEN / BC)
#define LOG2E    1.4426950408889634f

typedef __attribute__((ext_vector_type(16))) __bf16 bf16x16;
typedef __attribute__((ext_vector_type(8)))  __bf16 bf16x8;
typedef __attribute__((ext_vector_type(4)))  __bf16 bf16x4;
typedef __attribute__((ext_vector_type(2)))  __bf16 bf16x2;
typedef __attribute__((ext_vector_type(8)))  float  f32x8;
typedef __attribute__((ext_vector_type(4)))  float  f32x4;
typedef __attribute__((ext_vector_type(8)))  unsigned int u32x8;

static __device__ __forceinline__ unsigned int pack2(float a, float b) {
  bf16x2 t; t.x = (__bf16)a; t.y = (__bf16)b;
  return __builtin_bit_cast(unsigned int, t);
}
static __device__ __forceinline__ bf16x4 pack4(f32x4 v) {
  bf16x4 t; t.x = (__bf16)v.x; t.y = (__bf16)v.y;
  t.z = (__bf16)v.z; t.w = (__bf16)v.w;
  return t;
}
static __device__ __forceinline__ bf16x16 cat8(bf16x8 lo, bf16x8 hi) {
  return __builtin_shufflevector(lo, hi, 0,1,2,3,4,5,6,7,8,9,10,11,12,13,14,15);
}
// low 32 bits of a flat shared pointer == wave-relative LDS byte offset
static __device__ __forceinline__ unsigned lds_off(const void* p) {
  return (unsigned)(uintptr_t)p;
}
// CDNA5 async DMA: LDS[vdst + imm + b] = MEM[saddr + vaddr + imm + b], 16B/lane
static __device__ __forceinline__ void async_b128(unsigned lds_dst, unsigned voff,
                                                  uint64_t sbase, int imm) {
  asm volatile("global_load_async_to_lds_b128 %0, %1, %2 offset:%3"
               :: "v"(lds_dst), "v"(voff), "s"(sbase), "i"(imm) : "memory");
}
static __device__ __forceinline__ void wait_async() {
  asm volatile("s_wait_asynccnt 0x0" ::: "memory");
}

__global__ __launch_bounds__(NTHREADS, 1)
void fa1_wmma_kernel(const float* __restrict__ Qg, const float* __restrict__ Kg,
                     const float* __restrict__ Vg, float* __restrict__ Og)
{
  __shared__ __align__(16) __bf16 Qs [ROWS_WG * KS_STRIDE]; // Q x log2e, [row][d]
  __shared__ __align__(16) __bf16 Ksb[BC * KS_STRIDE];      // K tile bf16 [key][d]
  __shared__ __align__(16) __bf16 Vtb[HD * VT_STRIDE];      // V tile bf16 [d][key]
  __shared__ __align__(16) float  rawK[BC * HD];            // async staging (f32)
  __shared__ __align__(16) float  rawV[BC * HD];

  const int tid  = threadIdx.x;
  const int lane = tid & 31;
  const int wave = tid >> 5;
  const int lq   = lane & 15;
  const int uh   = lane >> 4;
  const int vkp  = (tid & 31) * 2;   // V conversion: key pair
  const int vd0  = (tid >> 5) * 32;  // V conversion: d range

  const unsigned v16   = (unsigned)tid * 16u;       // per-lane 16B slot
  const unsigned ldsK  = lds_off(rawK) + v16;
  const unsigned ldsV  = lds_off(rawV) + v16;

  // ---- kick off async DMA for tile 0, overlap with Q conversion
  {
    uint64_t kb = (uint64_t)(uintptr_t)Kg;
    uint64_t vb = (uint64_t)(uintptr_t)Vg;
    #pragma unroll
    for (int i = 0; i < 16; ++i) {
      async_b128(ldsK, v16, kb, i * 4096);
      async_b128(ldsV, v16, vb, i * 4096);
    }
  }
  // ---- Q -> LDS bf16 (pre-scaled by log2 e), flat-coalesced
  {
    const float* qsrc = Qg + (size_t)blockIdx.x * ROWS_WG * HD;
    #pragma unroll
    for (int i = 0; i < 32; ++i) {
      int f = tid + i * NTHREADS;
      int row = f >> 6;
      int d = (f & 63) * 4;
      f32x4 v = *(const f32x4*)(qsrc + (size_t)f * 4);
      v.x *= LOG2E; v.y *= LOG2E; v.z *= LOG2E; v.w *= LOG2E;
      *(bf16x4*)(Qs + row * KS_STRIDE + d) = pack4(v);
    }
  }
  wait_async();
  __syncthreads();
  // ---- convert tile 0
  {
    #pragma unroll
    for (int i = 0; i < 16; ++i) {
      int f = tid + i * NTHREADS;
      f32x4 v = *(const f32x4*)(rawK + (size_t)f * 4);
      *(bf16x4*)(&Ksb[(f >> 6) * KS_STRIDE + (f & 63) * 4]) = pack4(v);
    }
    const float* s0 = rawV + (size_t)vkp * HD + vd0;
    const float* s1 = s0 + HD;
    #pragma unroll
    for (int i = 0; i < 8; ++i) {
      f32x4 a = *(const f32x4*)(s0 + i * 4);
      f32x4 b = *(const f32x4*)(s1 + i * 4);
      #pragma unroll
      for (int j = 0; j < 4; ++j)
        *(bf16x2*)(&Vtb[(vd0 + i * 4 + j) * VT_STRIDE + vkp]) =
            __builtin_bit_cast(bf16x2, pack2(a[j], b[j]));
    }
  }
  __syncthreads();

  f32x8 acc[16];
  #pragma unroll
  for (int t = 0; t < 16; ++t) acc[t] = (f32x8)0.0f;
  float m_run = -INFINITY;
  float l_run = 0.0f;

  const __bf16* qrow = Qs + (wave * 16 + lq) * KS_STRIDE + uh * 16;

  for (int it = 0; it < NITER; ++it) {
    const bool pf = (it + 1 < NITER);

    // ---- issue async DMA for tile it+1 (no data VGPRs; runs under WMMAs)
    if (pf) {
      uint64_t kb = (uint64_t)(uintptr_t)(Kg + (size_t)(it + 1) * BC * HD);
      uint64_t vb = (uint64_t)(uintptr_t)(Vg + (size_t)(it + 1) * BC * HD);
      #pragma unroll
      for (int i = 0; i < 16; ++i) {
        async_b128(ldsK, v16, kb, i * 4096);
        async_b128(ldsV, v16, vb, i * 4096);
      }
    }

    // ---- S^T = K_tile x Q^T : 4 tiles of (16 keys x 16 queries)
    f32x8 st[4];
    #pragma unroll
    for (int kt = 0; kt < 4; ++kt) st[kt] = (f32x8)0.0f;
    #pragma unroll
    for (int c = 0; c < 8; ++c) {
      bf16x16 qc = *(const bf16x16*)(qrow + c * 32);   // contiguous B fragment
      #pragma unroll
      for (int kt = 0; kt < 4; ++kt) {
        const __bf16* arow = &Ksb[(kt * 16 + lq) * KS_STRIDE + uh * 16 + c * 32];
        bf16x16 a = cat8(*(const bf16x8*)(arow), *(const bf16x8*)(arow + 8));
        st[kt] = __builtin_amdgcn_wmma_f32_16x16x32_bf16(
            false, a, false, qc, (short)0, st[kt], false, false);
      }
    }

    // ---- online softmax (scores already in log2e units)
    float mloc = st[0][0];
    #pragma unroll
    for (int kt = 0; kt < 4; ++kt)
      #pragma unroll
      for (int r = 0; r < 8; ++r) mloc = fmaxf(mloc, st[kt][r]);
    mloc = fmaxf(mloc, __shfl_xor(mloc, 16, 32));
    const float m_new = fmaxf(m_run, mloc);
    const float scale = exp2f(m_run - m_new);
    m_run = m_new;

    float ls = 0.0f;
    unsigned int pk[4][4];
    #pragma unroll
    for (int kt = 0; kt < 4; ++kt) {
      #pragma unroll
      for (int j = 0; j < 4; ++j) {
        float p0 = exp2f(st[kt][2 * j]     - m_new);
        float p1 = exp2f(st[kt][2 * j + 1] - m_new);
        ls += p0 + p1;
        pk[kt][j] = pack2(p0, p1);
      }
    }
    l_run = l_run * scale + ls;

    // ---- assemble P^T B operands with one cross-half swap
    bf16x16 bp[2];
    #pragma unroll
    for (int h = 0; h < 2; ++h) {
      u32x8 w;
      #pragma unroll
      for (int j = 0; j < 4; ++j) {
        unsigned int lo_t = pk[2 * h][j];
        unsigned int hi_t = pk[2 * h + 1][j];
        unsigned int src  = uh ? lo_t : hi_t;
        unsigned int got  = __shfl_xor(src, 16, 32);
        w[j]     = uh ? got  : lo_t;
        w[4 + j] = uh ? hi_t : got;
      }
      bp[h] = __builtin_bit_cast(bf16x16, w);
    }

    // ---- rescale accumulator (skip when the running max didn't move)
    if (__any(scale != 1.0f)) {
      #pragma unroll
      for (int t = 0; t < 16; ++t)
        #pragma unroll
        for (int r = 0; r < 8; ++r) acc[t][r] *= scale;
    }

    // ---- O^T += V^T x P^T
    #pragma unroll
    for (int t = 0; t < 16; ++t) {
      const __bf16* vrow = &Vtb[(t * 16 + lq) * VT_STRIDE + uh * 16];
      #pragma unroll
      for (int h = 0; h < 2; ++h) {
        bf16x16 a = cat8(*(const bf16x8*)(vrow + h * 32),
                         *(const bf16x8*)(vrow + h * 32 + 8));
        acc[t] = __builtin_amdgcn_wmma_f32_16x16x32_bf16(
            false, a, false, bp[h], (short)0, acc[t], false, false);
      }
    }

    // ---- drain DMA, then convert raw f32 -> bf16 buffers for it+1
    wait_async();
    __syncthreads();            // compute reads done + DMA visible to all
    if (pf) {
      #pragma unroll
      for (int i = 0; i < 16; ++i) {
        int f = tid + i * NTHREADS;
        f32x4 v = *(const f32x4*)(rawK + (size_t)f * 4);
        *(bf16x4*)(&Ksb[(f >> 6) * KS_STRIDE + (f & 63) * 4]) = pack4(v);
      }
      const float* s0 = rawV + (size_t)vkp * HD + vd0;
      const float* s1 = s0 + HD;
      #pragma unroll
      for (int i = 0; i < 8; ++i) {
        f32x4 a = *(const f32x4*)(s0 + i * 4);
        f32x4 b = *(const f32x4*)(s1 + i * 4);
        #pragma unroll
        for (int j = 0; j < 4; ++j)
          *(bf16x2*)(&Vtb[(vd0 + i * 4 + j) * VT_STRIDE + vkp]) =
              __builtin_bit_cast(bf16x2, pack2(a[j], b[j]));
      }
    }
    __syncthreads();            // conversions visible before next compute
  }

  // ---- epilogue: merge l across halves, normalize, store O
  float l_tot = l_run + __shfl_xor(l_run, 16, 32);
  float inv = 1.0f / l_tot;
  float* orow = Og + (size_t)(blockIdx.x * ROWS_WG + wave * 16 + lq) * HD + uh * 8;
  #pragma unroll
  for (int t = 0; t < 16; ++t) {
    f32x4 o0, o1;
    o0.x = acc[t][0] * inv; o0.y = acc[t][1] * inv;
    o0.z = acc[t][2] * inv; o0.w = acc[t][3] * inv;
    o1.x = acc[t][4] * inv; o1.y = acc[t][5] * inv;
    o1.z = acc[t][6] * inv; o1.w = acc[t][7] * inv;
    *(f32x4*)(orow + t * 16)     = o0;
    *(f32x4*)(orow + t * 16 + 4) = o1;
  }
}

extern "C" void kernel_launch(void* const* d_in, const int* in_sizes, int n_in,
                              void* d_out, int out_size, void* d_ws, size_t ws_size,
                              hipStream_t stream) {
  (void)in_sizes; (void)n_in; (void)out_size; (void)d_ws; (void)ws_size;
  const float* Q = (const float*)d_in[0];
  const float* K = (const float*)d_in[1];
  const float* V = (const float*)d_in[2];
  float* O = (float*)d_out;
  dim3 grid(S_LEN / ROWS_WG);   // 256 workgroups x 128 query rows
  dim3 block(NTHREADS);         // 8 wave32 waves
  fa1_wmma_kernel<<<grid, block, 0, stream>>>(Q, K, V, O);
}
